// HeteroGNN_1288490189326
// MI455X (gfx1250) — compile-verified, matrix operand order checked by reading, not run on hardware
//
#include <hip/hip_runtime.h>

// ---------------- problem constants (from reference) ----------------
static constexpr int NRr  = 16384;   // reactions
static constexpr int NPp  = 40000;   // proteins
static constexpr int NMm  = 30000;   // molecules
static constexpr int Ee   = 200000;  // edges per edge type
static constexpr int EMBe = 1024;
static constexpr int HIDh = 256;
static constexpr int OUTo = 128;

// ---------------- GEMM tiling ----------------
static constexpr int BM = 128;
static constexpr int BN = 128;
static constexpr int BK = 32;

typedef __attribute__((ext_vector_type(16))) __bf16 bf16x16;
typedef __attribute__((ext_vector_type(8)))  float  f32x8;

// ---------------------------------------------------------------
// CDNA5 async global->LDS copy (ASYNCcnt-tracked), inline asm per
// CDNA5 ISA (GV mode: 64-bit VGPR address).
// ---------------------------------------------------------------
__device__ __forceinline__ void async_to_lds_b128(__bf16* lptr, const __bf16* gptr) {
    unsigned lds_off = (unsigned)(unsigned long long)lptr;          // low 32 bits = LDS offset
    unsigned long long ga = (unsigned long long)gptr;
    asm volatile("global_load_async_to_lds_b128 %0, %1, off"
                 :: "v"(lds_off), "v"(ga)
                 : "memory");
}
__device__ __forceinline__ void wait_async_all() {
    asm volatile("s_wait_asynccnt 0" ::: "memory");
}

// ---------------------------------------------------------------
// Edge-degree count (fp32 so we can reuse buffer as reciprocal)
// ---------------------------------------------------------------
__global__ void count_kernel(const int* __restrict__ dst, int ne, float* __restrict__ cnt) {
    int e = blockIdx.x * blockDim.x + threadIdx.x;
    if (e < ne) unsafeAtomicAdd(&cnt[dst[e]], 1.0f);
}

__global__ void invert_kernel(float* __restrict__ c, int n) {
    int i = blockIdx.x * blockDim.x + threadIdx.x;
    if (i < n) c[i] = 1.0f / fmaxf(c[i], 1.0f);
}

// ---------------------------------------------------------------
// fp32 [K,N] -> bf16 [N,K] (transposed) weight conversion, per matrix.
// Coalesced reads; scattered one-time writes.
// ---------------------------------------------------------------
__global__ void cvt_transpose_kernel(const float* __restrict__ in, __bf16* __restrict__ out,
                                     int K, int N, int nmat) {
    int idx = blockIdx.x * blockDim.x + threadIdx.x;
    int total = K * N;
    if (idx >= total * nmat) return;
    int mat = idx / total;
    int rem = idx - mat * total;
    int k = rem / N;
    int n = rem - k * N;
    out[(size_t)mat * total + (size_t)n * K + k] = (__bf16)in[idx];
}

// ---------------------------------------------------------------
// Edge scatter-add: acc[dst[e], :] += x[src[e], :]
// grid = (E, d/256), block = 256 -> coalesced along feature dim,
// hardware global_atomic_add_f32 via unsafeAtomicAdd.
// ---------------------------------------------------------------
__global__ __launch_bounds__(256)
void scatter_add_kernel(const float* __restrict__ x, const int* __restrict__ src,
                        const int* __restrict__ dst, float* __restrict__ acc, int d) {
    int e   = blockIdx.x;
    int col = blockIdx.y * 256 + threadIdx.x;
    int s = src[e];
    int t = dst[e];
    float v = x[(size_t)s * d + col];
    unsafeAtomicAdd(&acc[(size_t)t * d + col], v);
}

// ---------------------------------------------------------------
// WMMA GEMM: C[M,N] (+)= (rowScale .* A_f32 -> bf16) @ Bt_bf16 (+ bias)
// Bt is pre-transposed [N,K].
// Block: 256 threads = 8 waves; block tile 128x128; wave tile 32x64;
// 2x4 v_wmma_f32_16x16x32_bf16 accumulators per wave.
// A staged via VGPRs (fp32->bf16 + 1/deg scale); B staged via
// global_load_async_to_lds_b128 (ASYNCcnt) straight into LDS.
// ---------------------------------------------------------------
__global__ __launch_bounds__(256)
void gemm_bf16_wmma(const float* __restrict__ A, const float* __restrict__ rowScale,
                    const __bf16* __restrict__ Bt, const float* __restrict__ bias,
                    float* __restrict__ C, int M, int N, int K, int accumulate) {
    __shared__ __bf16 As[BM][BK + 8];   // [m][k], 80B row stride (16B aligned)
    __shared__ __bf16 Bs[BN][BK + 8];   // [n][k], same layout as As

    const int tid  = threadIdx.x;
    const int lane = tid & 31;
    const int wave = tid >> 5;
    const int wm   = (wave >> 1) * 32;   // wave M offset within block tile
    const int wn   = (wave & 1) * 64;    // wave N offset within block tile
    const int row0 = blockIdx.x * BM;
    const int col0 = blockIdx.y * BN;

    f32x8 c[2][4] = {};

    // A-tile fill mapping: 2 threads per row, 16 consecutive K each
    const int ar  = tid >> 1;
    const int ac0 = (tid & 1) * 16;
    // B-tile fill mapping: 512 16-byte chunks, 2 per thread
    const int bn0 = tid >> 2;            // 0..63  (rows tid>>2 and (tid+256)>>2)
    const int bp0 = tid & 3;             // 16B part within a 64B row

    // hoisted, branchless A row handling
    const int  gm   = row0 + ar;
    const bool ok   = gm < M;
    const int  gmc  = ok ? gm : 0;
    float scale     = ok ? (rowScale ? rowScale[gm] : 1.0f) : 0.0f;
    const float* aprow = A + (size_t)gmc * K;

    const int mlo = lane & 15;
    const int khi = (lane >> 4) << 3;    // 0 or 8 (K half select per ISA layout)

    for (int kk = 0; kk < K; kk += BK) {
        // ---- issue async B tile loads (bf16 [N,K] -> LDS [n][k]) ----
        {
            const __bf16* g0 = Bt + (size_t)(col0 + bn0) * K + kk + bp0 * 8;
            async_to_lds_b128(&Bs[bn0][bp0 * 8], g0);
            const __bf16* g1 = Bt + (size_t)(col0 + 64 + bn0) * K + kk + bp0 * 8;
            async_to_lds_b128(&Bs[64 + bn0][bp0 * 8], g1);
        }
        // ---- stage A tile (fp32 -> bf16, per-row scale folded in) ----
        {
            const float4* ap = reinterpret_cast<const float4*>(aprow + kk + ac0);
            if (kk + BK < K) __builtin_prefetch(aprow + kk + BK + ac0, 0, 1);
            __bf16* arow = &As[ar][ac0];
#pragma unroll
            for (int q = 0; q < 4; ++q) {
                float4 f = ap[q];
                arow[q * 4 + 0] = (__bf16)(f.x * scale);
                arow[q * 4 + 1] = (__bf16)(f.y * scale);
                arow[q * 4 + 2] = (__bf16)(f.z * scale);
                arow[q * 4 + 3] = (__bf16)(f.w * scale);
            }
        }
        wait_async_all();      // this wave's async LDS writes done
        __syncthreads();       // all waves' tiles visible

        // ---- build fragments per ISA 16-bit A/B layout ----
        bf16x16 a[2], b[4];
#pragma unroll
        for (int i = 0; i < 2; ++i) {
            int m = wm + i * 16 + mlo;
#pragma unroll
            for (int h = 0; h < 16; ++h) {
                int v = h >> 1;
                int k = ((v & 4) << 2) + khi + ((v & 3) << 1) + (h & 1);
                a[i][h] = As[m][k];
            }
        }
#pragma unroll
        for (int j = 0; j < 4; ++j) {
            int n = wn + j * 16 + mlo;
#pragma unroll
            for (int h = 0; h < 16; ++h) {
                int v = h >> 1;
                int k = ((v & 4) << 2) + khi + ((v & 3) << 1) + (h & 1);
                b[j][h] = Bs[n][k];
            }
        }

        // ---- 8 WMMAs per K-step ----
#pragma unroll
        for (int i = 0; i < 2; ++i)
#pragma unroll
            for (int j = 0; j < 4; ++j)
                c[i][j] = __builtin_amdgcn_wmma_f32_16x16x32_bf16(
                    false, a[i], false, b[j], (short)0, c[i][j], false, false);

        __syncthreads();
    }

    // ---- epilogue: C/D layout lane mapping (VGPR r -> row r / r+8) ----
#pragma unroll
    for (int i = 0; i < 2; ++i) {
#pragma unroll
        for (int r = 0; r < 8; ++r) {
            int m = row0 + wm + i * 16 + ((lane >> 4) << 3) + r;
            if (m < M) {
#pragma unroll
                for (int j = 0; j < 4; ++j) {
                    int n = col0 + wn + j * 16 + (lane & 15);
                    float v = c[i][j][r];
                    if (bias) v += bias[n];
                    size_t idx = (size_t)m * N + n;
                    if (accumulate) v += C[idx];
                    C[idx] = v;
                }
            }
        }
    }
}

// ---------------------------------------------------------------
// Cosine similarity head: one wave32 per reaction row
// ---------------------------------------------------------------
__global__ __launch_bounds__(256)
void cosine_kernel(const float* __restrict__ ro, const float* __restrict__ projs,
                   const int* __restrict__ types, float* __restrict__ out, int nr) {
    int row  = blockIdx.x * 8 + (threadIdx.x >> 5);
    int lane = threadIdx.x & 31;
    if (row >= nr) return;
    int t = types[row];
    const float* a = ro + (size_t)row * OUTo;
    const float* b = projs + ((size_t)t * nr + row) * OUTo;
    float dot = 0.f, na = 0.f, nb = 0.f;
    for (int k = lane; k < OUTo; k += 32) {
        float av = a[k], bv = b[k];
        dot += av * bv; na += av * av; nb += bv * bv;
    }
    for (int o = 16; o; o >>= 1) {
        dot += __shfl_xor(dot, o, 32);
        na  += __shfl_xor(na,  o, 32);
        nb  += __shfl_xor(nb,  o, 32);
    }
    if (lane == 0) {
        float den = fmaxf(sqrtf(na) * sqrtf(nb), 1e-8f);
        out[row] = (dot / den + 1.0f) * 0.5f;
    }
}

// ---------------------------------------------------------------
extern "C" void kernel_launch(void* const* d_in, const int* in_sizes, int n_in,
                              void* d_out, int out_size, void* d_ws, size_t ws_size,
                              hipStream_t stream) {
    const float* x_r   = (const float*)d_in[0];
    const float* x_p   = (const float*)d_in[1];
    const float* x_m   = (const float*)d_in[2];
    const int* p2r_s = (const int*)d_in[3];
    const int* p2r_d = (const int*)d_in[4];
    const int* m2r_s = (const int*)d_in[5];
    const int* m2r_d = (const int*)d_in[6];
    const int* r2p_s = (const int*)d_in[7];
    const int* r2p_d = (const int*)d_in[8];
    const int* r2m_s = (const int*)d_in[9];
    const int* r2m_d = (const int*)d_in[10];
    const float* notes = (const float*)d_in[11];
    const int*   types = (const int*)d_in[12];
    const float* Wl0 = (const float*)d_in[13];
    const float* Wr0 = (const float*)d_in[14];
    const float* bl0 = (const float*)d_in[15];
    const float* Wl  = (const float*)d_in[16];
    const float* Wr  = (const float*)d_in[17];
    const float* bl  = (const float*)d_in[18];
    const float* Wre = (const float*)d_in[19];
    const float* bre = (const float*)d_in[20];
    const float* Wno = (const float*)d_in[21];
    const float* bno = (const float*)d_in[22];

    // ---- workspace carve-out (256B aligned slabs) ----
    char* ws = (char*)d_ws;
    size_t off = 0;
    auto take = [&](size_t bytes) -> char* {
        char* p = ws + off;
        off = (off + bytes + 255) & ~(size_t)255;
        return p;
    };
    float* inv_p2r = (float*)take((size_t)NRr * 4);
    float* inv_m2r = (float*)take((size_t)NRr * 4);
    float* inv_r2p = (float*)take((size_t)NPp * 4);
    float* inv_r2m = (float*)take((size_t)NMm * 4);
    float* acc     = (float*)take((size_t)NPp * EMBe * 4);   // reused for every edge type
    float* xrA = (float*)take((size_t)NRr * HIDh * 4);
    float* xpA = (float*)take((size_t)NPp * HIDh * 4);
    float* xmA = (float*)take((size_t)NMm * HIDh * 4);
    float* xrB = (float*)take((size_t)NRr * HIDh * 4);
    float* xpB = (float*)take((size_t)NPp * HIDh * 4);
    float* xmB = (float*)take((size_t)NMm * HIDh * 4);
    __bf16* wl0b = (__bf16*)take((size_t)4 * EMBe * HIDh * 2);   // transposed [N,K] per matrix
    __bf16* wr0b = (__bf16*)take((size_t)4 * EMBe * HIDh * 2);
    __bf16* wlb  = (__bf16*)take((size_t)3 * 4 * HIDh * HIDh * 2);
    __bf16* wrb  = (__bf16*)take((size_t)3 * 4 * HIDh * HIDh * 2);
    __bf16* wreb = (__bf16*)take((size_t)HIDh * OUTo * 2);
    __bf16* wnob = (__bf16*)take((size_t)3 * EMBe * OUTo * 2);
    float* ro    = (float*)take((size_t)NRr * OUTo * 4);
    float* projs = (float*)take((size_t)3 * NRr * OUTo * 4);

    // ---- edge-degree reciprocals (edges are fixed across layers) ----
    hipMemsetAsync(inv_p2r, 0, (size_t)NRr * 4, stream);
    hipMemsetAsync(inv_m2r, 0, (size_t)NRr * 4, stream);
    hipMemsetAsync(inv_r2p, 0, (size_t)NPp * 4, stream);
    hipMemsetAsync(inv_r2m, 0, (size_t)NMm * 4, stream);
    int cg = (Ee + 255) / 256;
    count_kernel<<<cg, 256, 0, stream>>>(p2r_d, Ee, inv_p2r);
    count_kernel<<<cg, 256, 0, stream>>>(m2r_d, Ee, inv_m2r);
    count_kernel<<<cg, 256, 0, stream>>>(r2p_d, Ee, inv_r2p);
    count_kernel<<<cg, 256, 0, stream>>>(r2m_d, Ee, inv_r2m);
    invert_kernel<<<(NRr + 255) / 256, 256, 0, stream>>>(inv_p2r, NRr);
    invert_kernel<<<(NRr + 255) / 256, 256, 0, stream>>>(inv_m2r, NRr);
    invert_kernel<<<(NPp + 255) / 256, 256, 0, stream>>>(inv_r2p, NPp);
    invert_kernel<<<(NMm + 255) / 256, 256, 0, stream>>>(inv_r2m, NMm);

    // ---- weights -> bf16, transposed [N,K], once per launch ----
    auto cvtT = [&](const float* src, __bf16* dst, int K, int N, int nmat) {
        size_t n = (size_t)K * N * nmat;
        cvt_transpose_kernel<<<(int)((n + 255) / 256), 256, 0, stream>>>(src, dst, K, N, nmat);
    };
    cvtT(Wl0, wl0b, EMBe, HIDh, 4);
    cvtT(Wr0, wr0b, EMBe, HIDh, 4);
    cvtT(Wl,  wlb,  HIDh, HIDh, 12);
    cvtT(Wr,  wrb,  HIDh, HIDh, 12);
    cvtT(Wre, wreb, HIDh, OUTo, 1);
    cvtT(Wno, wnob, EMBe, OUTo, 3);

    auto gemm = [&](const float* A, const float* scale, const __bf16* B, const float* bias,
                    float* C, int M, int N, int K, int accum) {
        dim3 g((M + BM - 1) / BM, N / BN);
        gemm_bf16_wmma<<<g, 256, 0, stream>>>(A, scale, B, bias, C, M, N, K, accum);
    };
    auto scatter = [&](const float* x, const int* s, const int* d, float* a, int dd) {
        dim3 g(Ee, dd / 256);
        scatter_add_kernel<<<g, 256, 0, stream>>>(x, s, d, a, dd);
    };

    const float *cxr = x_r, *cxp = x_p, *cxm = x_m;
    float *nxr = xrA, *nxp = xpA, *nxm = xmA;
    float *oxr = xrB, *oxp = xpB, *oxm = xmB;

    for (int l = 0; l < 4; ++l) {
        int d = (l == 0) ? EMBe : HIDh;
        const __bf16* WlB = (l == 0) ? wl0b : wlb + (size_t)(l - 1) * 4 * HIDh * HIDh;
        const __bf16* WrB = (l == 0) ? wr0b : wrb + (size_t)(l - 1) * 4 * HIDh * HIDh;
        const float*  blF = (l == 0) ? bl0  : bl  + (size_t)(l - 1) * 4 * HIDh;
        size_t wstride = (size_t)d * HIDh;   // per-matrix element stride (same transposed)

        // reaction <- p2r + m2r + 2x self
        hipMemsetAsync(acc, 0, (size_t)NRr * d * 4, stream);
        scatter(cxp, p2r_s, p2r_d, acc, d);
        gemm(acc, inv_p2r, WlB + 0 * wstride, blF + 0 * HIDh, nxr, NRr, HIDh, d, 0);
        gemm(cxr, nullptr, WrB + 0 * wstride, nullptr,        nxr, NRr, HIDh, d, 1);
        hipMemsetAsync(acc, 0, (size_t)NRr * d * 4, stream);
        scatter(cxm, m2r_s, m2r_d, acc, d);
        gemm(acc, inv_m2r, WlB + 1 * wstride, blF + 1 * HIDh, nxr, NRr, HIDh, d, 1);
        gemm(cxr, nullptr, WrB + 1 * wstride, nullptr,        nxr, NRr, HIDh, d, 1);

        // protein <- r2p + self
        hipMemsetAsync(acc, 0, (size_t)NPp * d * 4, stream);
        scatter(cxr, r2p_s, r2p_d, acc, d);
        gemm(acc, inv_r2p, WlB + 2 * wstride, blF + 2 * HIDh, nxp, NPp, HIDh, d, 0);
        gemm(cxp, nullptr, WrB + 2 * wstride, nullptr,        nxp, NPp, HIDh, d, 1);

        // molecule <- r2m + self
        hipMemsetAsync(acc, 0, (size_t)NMm * d * 4, stream);
        scatter(cxr, r2m_s, r2m_d, acc, d);
        gemm(acc, inv_r2m, WlB + 3 * wstride, blF + 3 * HIDh, nxm, NMm, HIDh, d, 0);
        gemm(cxm, nullptr, WrB + 3 * wstride, nullptr,        nxm, NMm, HIDh, d, 1);

        // ping-pong
        cxr = nxr; cxp = nxp; cxm = nxm;
        float* t;
        t = nxr; nxr = oxr; oxr = t;
        t = nxp; nxp = oxp; oxp = t;
        t = nxm; nxm = oxm; oxm = t;
    }

    // ---- heads ----
    gemm(cxr, nullptr, wreb, bre, ro, NRr, OUTo, HIDh, 0);
    for (int t = 0; t < 3; ++t)
        gemm(notes, nullptr, wnob + (size_t)t * EMBe * OUTo, bno + t * OUTo,
             projs + (size_t)t * NRr * OUTo, NRr, OUTo, EMBe, 0);

    cosine_kernel<<<(NRr + 7) / 8, 256, 0, stream>>>(ro, projs, types, (float*)d_out, NRr);
}
